// MultiHeadAttention_7430293422727
// MI455X (gfx1250) — matrix operand order repaired and checked by reading, not compiled, not run on hardware
//
#include <hip/hip_runtime.h>
#include <hip/hip_bf16.h>

typedef __attribute__((ext_vector_type(16))) __bf16 v16bf;
typedef __attribute__((ext_vector_type(8)))  __bf16 v8bf;
typedef __attribute__((ext_vector_type(8)))  float  v8f;

#define BATCH 4
#define SEQ   2048
#define DMODEL 1024
#define HEADS 16
#define DHEAD 64
#define MASK_CONST 4294967296.0f
#define ATTN_SCALE 0.125f   // 1/sqrt(64)

union V16U { v16bf v; v8bf h[2]; };

// ---------------------------------------------------------------------------
// fp32 -> bf16 elementwise conversion (activations)
// ---------------------------------------------------------------------------
__global__ void cvt_f32_bf16_kernel(const float* __restrict__ src,
                                    __bf16* __restrict__ dst, int n) {
    int i = blockIdx.x * blockDim.x + threadIdx.x;
    if (i < n) dst[i] = (__bf16)src[i];
}

// ---------------------------------------------------------------------------
// Pack a 1024x1024 fp32 weight into bf16 B-fragment order:
//   P[frag*512 + lane*16 + e], frag = (kb/32)*64 + ntile,
//   element e of lane <-> (k = kb*32 + (lane>>4)*16 + e, n = ntile*16 + (lane&15))
// One lane's fragment = 32 contiguous bytes -> 2x global_load_b128 in the GEMM.
// ---------------------------------------------------------------------------
__global__ void pack_weight_kernel(const float* __restrict__ W,
                                   __bf16* __restrict__ P) {
    int idx  = blockIdx.x * blockDim.x + threadIdx.x;  // 0 .. 1048575
    int e    = idx & 15;
    int lane = (idx >> 4) & 31;
    int frag = idx >> 9;
    int ntile = frag & 63;
    int kb    = frag >> 6;
    int half = lane >> 4, l15 = lane & 15;
    int k = kb * 32 + half * 16 + e;
    int n = ntile * 16 + l15;
    P[idx] = (__bf16)W[(size_t)k * DMODEL + n];
}

// ---------------------------------------------------------------------------
// Wave32 row reductions across the 16 lanes that hold one C-matrix row
// ---------------------------------------------------------------------------
__device__ __forceinline__ float rowmax16(float x) {
#pragma unroll
    for (int off = 8; off > 0; off >>= 1) x = fmaxf(x, __shfl_xor(x, off, 16));
    return x;
}
__device__ __forceinline__ float rowsum16(float x) {
#pragma unroll
    for (int off = 8; off > 0; off >>= 1) x += __shfl_xor(x, off, 16);
    return x;
}

// ---------------------------------------------------------------------------
// bf16 WMMA GEMM:  C[8192,1024] = A[8192,1024] * W[1024,1024] + bias
//   A: bf16 row-major, staged to LDS in natural k order so an
//      A-fragment is one contiguous 32B run per lane (2x ds_load_b128).
//   B: pre-packed fragment-order bf16 in global (L2-resident), no LDS.
//   Block 128 threads = 4 waves; block tile 128x64; wave tile 32x64
//   (2 A-fragments x 4 N-tiles = 8 WMMA per 32-deep k-step).
//   mode 0: bf16 out, head-split (b,h,s,d)             (Q, K)
//   mode 2: bf16 out, head-split transposed (b,h,d,s)  (V)
//   mode 1: fp32 out, row-major (m,n) + bias           (final projection)
// ---------------------------------------------------------------------------
__global__ void gemm_bf16_kernel(const __bf16* __restrict__ A,
                                 const __bf16* __restrict__ Pw,
                                 const float*  __restrict__ bias,
                                 __bf16* __restrict__ outBf,
                                 float*  __restrict__ outF,
                                 int mode) {
    __shared__ __bf16 ldsA[128 * 32];   // [row][k]

    const int tid  = threadIdx.x;
    const int lane = tid & 31;
    const int wave = tid >> 5;
    const int half = lane >> 4;
    const int l15  = lane & 15;
    const int blockM = blockIdx.x * 128;
    const int blockN = blockIdx.y * 64;
    const int waveM  = wave * 32;

    v8f acc[8] = {};

    for (int kb = 0; kb < DMODEL; kb += 32) {
        // --- stage A 128x32 (512 x 16B), natural 8-elem chunk order ---
#pragma unroll
        for (int u = 0; u < 4; ++u) {
            int v = tid + u * 128;
            int row = v >> 2, kv = v & 3;
            int pos = kv;
            reinterpret_cast<uint4*>(ldsA)[row * 4 + pos] =
                *reinterpret_cast<const uint4*>(A + (size_t)(blockM + row) * DMODEL + kb + kv * 8);
        }
        __syncthreads();

        // Two A fragments (rows waveM.. and waveM+16..), contiguous 32B each
        v16bf af0 = *reinterpret_cast<const v16bf*>(&ldsA[(waveM + l15) * 32 + half * 16]);
        v16bf af1 = *reinterpret_cast<const v16bf*>(&ldsA[(waveM + 16 + l15) * 32 + half * 16]);

        // B fragments straight from packed global (L2); each feeds 2 WMMA
        const __bf16* pB = Pw + ((size_t)(kb >> 5) * 64 + (blockN >> 4)) * 512 + lane * 16;
#pragma unroll
        for (int t = 0; t < 4; ++t) {
            v16bf bf_ = *reinterpret_cast<const v16bf*>(pB + (size_t)t * 512);
            acc[t] = __builtin_amdgcn_wmma_f32_16x16x32_bf16(
                false, af0, false, bf_, (short)0, acc[t], false, false);
            acc[4 + t] = __builtin_amdgcn_wmma_f32_16x16x32_bf16(
                false, af1, false, bf_, (short)0, acc[4 + t], false, false);
        }
        __syncthreads();
    }

    // --- epilogue ---
#pragma unroll
    for (int rg = 0; rg < 2; ++rg) {
#pragma unroll
        for (int t = 0; t < 4; ++t) {
#pragma unroll
            for (int i = 0; i < 8; ++i) {
                int m = blockM + waveM + rg * 16 + i + half * 8;
                int n = blockN + t * 16 + l15;
                float val = acc[rg * 4 + t][i] + bias[n];
                if (mode == 1) {
                    outF[(size_t)m * DMODEL + n] = val;
                } else {
                    int b = m >> 11, s = m & (SEQ - 1);
                    int h = n >> 6, d = n & 63;
                    size_t bh = (size_t)(b * HEADS + h);
                    size_t idx = (mode == 0) ? (bh * SEQ + s) * DHEAD + d
                                             : (bh * DHEAD + d) * SEQ + s;   // mode 2: V^T
                    outBf[idx] = (__bf16)val;
                }
            }
        }
    }
}

// ---------------------------------------------------------------------------
// Flash attention.  Qh/Kh: (B*H, S, 64) bf16;  Vt: (B*H, 64, S) bf16.
// Block = 128 threads (4 waves); wave owns 16 query rows; 32-key chunks.
// All WMMA fragments are contiguous 32B per lane.
// ---------------------------------------------------------------------------
__global__ void flash_attn_kernel(const __bf16* __restrict__ Qh,
                                  const __bf16* __restrict__ Kh,
                                  const __bf16* __restrict__ Vt,
                                  __bf16* __restrict__ AO) {
    __shared__ __bf16 ldsK[32 * 64];        // [key][d]   (natural layout)
    __shared__ __bf16 ldsVT[64 * 32];       // [d][key]   (from V^T global)
    __shared__ __bf16 ldsP[4][16 * 32];     // [row][key]

    const int tid  = threadIdx.x;
    const int lane = tid & 31;
    const int wave = tid >> 5;
    const int half = lane >> 4;
    const int l15  = lane & 15;
    const int bh    = blockIdx.x >> 5;
    const int qtile = blockIdx.x & 31;
    const int b = bh >> 4, h = bh & 15;
    const int qrow0 = qtile * 64 + wave * 16;

    // --- Q tile: two 16x32 A-fragments via 4x global_load_b128 ---
    const __bf16* Qbase = Qh + ((size_t)bh * SEQ + qrow0) * DHEAD + (size_t)l15 * DHEAD;
    V16U qa0, qa1;
    qa0.h[0] = *reinterpret_cast<const v8bf*>(Qbase + half * 16);
    qa0.h[1] = *reinterpret_cast<const v8bf*>(Qbase + 8 + half * 16);
    qa1.h[0] = *reinterpret_cast<const v8bf*>(Qbase + 32 + half * 16);
    qa1.h[1] = *reinterpret_cast<const v8bf*>(Qbase + 40 + half * 16);

    float m_r[8], l_r[8];
    v8f o_acc[4] = {};
#pragma unroll
    for (int i = 0; i < 8; ++i) { m_r[i] = -__builtin_inff(); l_r[i] = 0.0f; }

    const __bf16* Kbase = Kh + (size_t)bh * SEQ * DHEAD;
    const __bf16* Vbase = Vt + (size_t)bh * DHEAD * SEQ;

    // P store positions: natural key order so reads are contiguous
    const int p0pos = l15;                                   // key = l15
    const int p1pos = 16 + p0pos;                            // key = 16 + l15

    for (int kc = 0; kc < SEQ; kc += 32) {
        // --- stage K chunk [32 keys][64 d] and V^T chunk [64 d][32 keys] ---
        {
            const uint4* gk = reinterpret_cast<const uint4*>(Kbase + (size_t)kc * DHEAD);
            uint4* sk = reinterpret_cast<uint4*>(ldsK);
            sk[tid]       = gk[tid];
            sk[tid + 128] = gk[tid + 128];
#pragma unroll
            for (int u = 0; u < 2; ++u) {
                int v = tid + u * 128;
                int dv = v >> 2, kvv = v & 3;
                reinterpret_cast<uint4*>(ldsVT)[v] =
                    *reinterpret_cast<const uint4*>(Vbase + (size_t)dv * SEQ + kc + kvv * 8);
            }
        }
        if (kc + 32 < SEQ) {   // speculative prefetch of next chunks
            __builtin_prefetch(Kbase + (size_t)(kc + 32) * DHEAD + lane * 32, 0, 0);
            __builtin_prefetch(Vbase + (size_t)(lane * 2) * SEQ + kc + 32, 0, 0);
        }
        __syncthreads();

        // --- scores: two 16-key tiles x two 32-deep k-steps ---
        v8f s[2] = {};
#pragma unroll
        for (int t = 0; t < 2; ++t) {
            const __bf16* kp = &ldsK[(t * 16 + l15) * 64 + half * 16];
            v16bf bk0 = *reinterpret_cast<const v16bf*>(kp);        // d 0..31
            s[t] = __builtin_amdgcn_wmma_f32_16x16x32_bf16(
                false, qa0.v, false, bk0, (short)0, s[t], false, false);
            v16bf bk1 = *reinterpret_cast<const v16bf*>(kp + 32);   // d 32..63
            s[t] = __builtin_amdgcn_wmma_f32_16x16x32_bf16(
                false, qa1.v, false, bk1, (short)0, s[t], false, false);
        }

        // --- mask + scale + online softmax (row i lives in one 16-lane half) ---
#pragma unroll
        for (int i = 0; i < 8; ++i) {
            int rowg = qrow0 + i + half * 8;
            float x0 = s[0][i] + (((kc + l15) > rowg) ? MASK_CONST : 0.0f);
            float x1 = s[1][i] + (((kc + 16 + l15) > rowg) ? MASK_CONST : 0.0f);
            x0 *= ATTN_SCALE;  x1 *= ATTN_SCALE;

            float cm = rowmax16(fmaxf(x0, x1));
            float mn = fmaxf(m_r[i], cm);
            float sf = __expf(m_r[i] - mn);
            float p0 = __expf(x0 - mn);
            float p1 = __expf(x1 - mn);
            float rs = rowsum16(p0 + p1);
            l_r[i] = l_r[i] * sf + rs;
            m_r[i] = mn;
#pragma unroll
            for (int j = 0; j < 4; ++j) o_acc[j][i] *= sf;

            ldsP[wave][(i + half * 8) * 32 + p0pos] = (__bf16)p0;
            ldsP[wave][(i + half * 8) * 32 + p1pos] = (__bf16)p1;
        }

        // --- P (16x32) A-fragment (contiguous), P·V over 4 dv tiles ---
        v16bf pf = *reinterpret_cast<const v16bf*>(&ldsP[wave][l15 * 32 + half * 16]);
#pragma unroll
        for (int j = 0; j < 4; ++j) {
            v16bf vf = *reinterpret_cast<const v16bf*>(&ldsVT[(j * 16 + l15) * 32 + half * 16]);
            o_acc[j] = __builtin_amdgcn_wmma_f32_16x16x32_bf16(
                false, pf, false, vf, (short)0, o_acc[j], false, false);
        }
        __syncthreads();
    }

    // --- normalize and emit (B, S, H*64) bf16 ---
#pragma unroll
    for (int j = 0; j < 4; ++j) {
#pragma unroll
        for (int i = 0; i < 8; ++i) {
            int srow = qrow0 + i + half * 8;
            float val = o_acc[j][i] / l_r[i];
            AO[((size_t)b * SEQ + srow) * DMODEL + h * DHEAD + j * 16 + l15] = (__bf16)val;
        }
    }
}

// ---------------------------------------------------------------------------
// launcher
// ---------------------------------------------------------------------------
extern "C" void kernel_launch(void* const* d_in, const int* in_sizes, int n_in,
                              void* d_out, int out_size, void* d_ws, size_t ws_size,
                              hipStream_t stream) {
    const float* queries = (const float*)d_in[0];
    const float* keys    = (const float*)d_in[1];
    const float* values  = (const float*)d_in[2];
    // d_in[3] = mask (causal triu, applied analytically)
    const float* Wq = (const float*)d_in[4];  const float* bq = (const float*)d_in[5];
    const float* Wk = (const float*)d_in[6];  const float* bk = (const float*)d_in[7];
    const float* Wv = (const float*)d_in[8];  const float* bv = (const float*)d_in[9];
    const float* Wo = (const float*)d_in[10]; const float* bo = (const float*)d_in[11];
    float* out = (float*)d_out;

    const size_t ACT = (size_t)BATCH * SEQ * DMODEL;   // 8388608
    const size_t WSZ = (size_t)DMODEL * DMODEL;        // 1048576

    char* ws = (char*)d_ws;  size_t off = 0;
    auto carve = [&](size_t bytes) { char* p = ws + off; off += (bytes + 255) & ~(size_t)255; return p; };
    __bf16* Xq  = (__bf16*)carve(ACT * 2);
    __bf16* Xk  = (__bf16*)carve(ACT * 2);
    __bf16* Xv  = (__bf16*)carve(ACT * 2);
    __bf16* Pq  = (__bf16*)carve(WSZ * 2);   // packed weights (fragment order)
    __bf16* Pk  = (__bf16*)carve(WSZ * 2);
    __bf16* Pv  = (__bf16*)carve(WSZ * 2);
    __bf16* Po  = (__bf16*)carve(WSZ * 2);
    __bf16* Qh  = (__bf16*)carve(ACT * 2);   // (B*H, S, 64)
    __bf16* Kh  = (__bf16*)carve(ACT * 2);   // (B*H, S, 64)
    __bf16* Vt  = (__bf16*)carve(ACT * 2);   // (B*H, 64, S)
    __bf16* AO  = (__bf16*)carve(ACT * 2);   // (B, S, H*64)

    // 1) convert activations, pack weights
    cvt_f32_bf16_kernel<<<(ACT + 255) / 256, 256, 0, stream>>>(queries, Xq, (int)ACT);
    cvt_f32_bf16_kernel<<<(ACT + 255) / 256, 256, 0, stream>>>(keys,    Xk, (int)ACT);
    cvt_f32_bf16_kernel<<<(ACT + 255) / 256, 256, 0, stream>>>(values,  Xv, (int)ACT);
    pack_weight_kernel<<<WSZ / 256, 256, 0, stream>>>(Wq, Pq);
    pack_weight_kernel<<<WSZ / 256, 256, 0, stream>>>(Wk, Pk);
    pack_weight_kernel<<<WSZ / 256, 256, 0, stream>>>(Wv, Pv);
    pack_weight_kernel<<<WSZ / 256, 256, 0, stream>>>(Wo, Po);

    // 2) projections
    dim3 ggrid((BATCH * SEQ) / 128, DMODEL / 64);
    gemm_bf16_kernel<<<ggrid, 128, 0, stream>>>(Xq, Pq, bq, Qh, nullptr, 0);
    gemm_bf16_kernel<<<ggrid, 128, 0, stream>>>(Xk, Pk, bk, Kh, nullptr, 0);
    gemm_bf16_kernel<<<ggrid, 128, 0, stream>>>(Xv, Pv, bv, Vt, nullptr, 2);

    // 3) attention
    flash_attn_kernel<<<BATCH * HEADS * (SEQ / 64), 128, 0, stream>>>(Qh, Kh, Vt, AO);

    // 4) output projection (fp32 + bias -> d_out)
    gemm_bf16_kernel<<<ggrid, 128, 0, stream>>>(AO, Po, bo, nullptr, out, 1);
}
